// DynamicGCN_27367531610736
// MI455X (gfx1250) — matrix operand order repaired and checked by reading, not compiled, not run on hardware
//
#include <hip/hip_runtime.h>
#include <hip/hip_bf16.h>

// ---------------- problem constants (match reference) ----------------
#define NN 50000      // nodes
#define NE 800000     // edges
#define NG 500        // graphs
#define DI 128
#define DH 128
#define DOUTF 64

typedef __attribute__((ext_vector_type(16))) __bf16 v16bf;
typedef __attribute__((ext_vector_type(8)))  float  v8f;

static __device__ __forceinline__ unsigned short f2bf(float f) {
    union { float f; unsigned u; } v; v.f = f;
    unsigned u = v.u;
    u += 0x7FFFu + ((u >> 16) & 1u);          // round-to-nearest-even
    return (unsigned short)(u >> 16);
}

// ---------------- utility: zero a float buffer ----------------
__global__ void k_zero(float* p, long n) {
    long i = (long)blockIdx.x * blockDim.x + threadIdx.x;
    if (i < n) p[i] = 0.0f;
}

// ---------------- degrees via f32 atomics ----------------
__global__ void k_degrees(const int* __restrict__ src, const int* __restrict__ dst,
                          float* degO, float* degI, int nE) {
    int e = blockIdx.x * blockDim.x + threadIdx.x;
    if (e < nE) {
        atomicAdd(&degO[src[e]], 1.0f);
        atomicAdd(&degI[dst[e]], 1.0f);
    }
}

// ---------------- deg -> norm (in place) ----------------
__global__ void k_norms(float* degO, float* degI, int n) {
    int i = blockIdx.x * blockDim.x + threadIdx.x;
    if (i < n) {
        float a = degO[i];
        degO[i] = (a > 0.0f) ? rsqrtf(fmaxf(a, 1.0f)) : 0.0f;
        float b = degI[i];
        degI[i] = (b > 0.0f) ? rsqrtf(fmaxf(b, 1.0f)) : 0.0f;
    }
}

// ---------------- f32 -> bf16 weight conversion ----------------
__global__ void k_cvt_bf16(const float* __restrict__ w, unsigned short* __restrict__ o, int n) {
    int i = blockIdx.x * blockDim.x + threadIdx.x;
    if (i < n) o[i] = f2bf(w[i]);
}

// ---------------- edge gather-scale-scatter: agg[dst] += h[src]*norm_out[src]
// one wave (32 lanes) per edge, 4 channels/lane (D = 128). The 25.6 MB feature
// matrix is L2-resident (192 MB L2), so gathers and scatter-atomics stay on-chip.
__global__ __launch_bounds__(256) void k_scatter(const float* __restrict__ h,
                                                 const float* __restrict__ normO,
                                                 const int* __restrict__ src,
                                                 const int* __restrict__ dst,
                                                 float* __restrict__ agg, int nE) {
    int e = blockIdx.x * 8 + (threadIdx.x >> 5);
    if (e >= nE) return;
    int lane = threadIdx.x & 31;
    int s = src[e];
    int d = dst[e];
    float w = normO[s];
    const float4* h4 = (const float4*)h;
    float4 v = h4[(size_t)s * 32 + lane];
    float* ap = agg + (size_t)d * 128 + lane * 4;
    atomicAdd(ap + 0, v.x * w);
    atomicAdd(ap + 1, v.y * w);
    atomicAdd(ap + 2, v.z * w);
    atomicAdd(ap + 3, v.w * w);
}

// ---------------- WMMA GEMM: out = relu((agg * norm_in) @ W + b) ----------------
// Block: 256 threads = 8 waves; 64 rows/block staged in LDS as bf16 (pre-scaled
// by norm_in). K = 128 fully resident. Each wave owns 16x16 output tiles and
// runs 4x V_WMMA_F32_16X16X32_BF16 along K. Fragment layouts per CDNA5 ISA 7.12.2.
// tn is loop-invariant per wave -> the compiler keeps B fragments in registers.
template <int DOUT>
__global__ __launch_bounds__(256) void k_gemm_relu(const float* __restrict__ agg,
                                                   const float* __restrict__ normI,
                                                   const unsigned short* __restrict__ Wb,
                                                   const float* __restrict__ bias,
                                                   float* __restrict__ out, int M) {
    __shared__ __bf16 sA[64 * 128];
    __shared__ __bf16 sW[128 * DOUT];
    unsigned short* sAu = reinterpret_cast<unsigned short*>(sA);
    unsigned short* sWu = reinterpret_cast<unsigned short*>(sW);

    int tid = threadIdx.x;
    int blockRow = blockIdx.x * 64;

    // stage bf16 weights (K-major, [128][DOUT])
    for (int i = tid; i < 128 * DOUT; i += 256) sWu[i] = Wb[i];

    // stage 64x128 activation tile, scaled by norm_in, converted to bf16
    const float4* a4 = (const float4*)agg;
    #pragma unroll
    for (int j = 0; j < 8; ++j) {
        int l = j * 256 + tid;            // over 64*32 float4s
        int row = l >> 5;
        int c4  = l & 31;
        int grow = blockRow + row;
        float4 v = make_float4(0.f, 0.f, 0.f, 0.f);
        float w = 0.f;
        if (grow < M) { w = normI[grow]; v = a4[(size_t)grow * 32 + c4]; }
        int base = row * 128 + c4 * 4;
        sAu[base + 0] = f2bf(v.x * w);
        sAu[base + 1] = f2bf(v.y * w);
        sAu[base + 2] = f2bf(v.z * w);
        sAu[base + 3] = f2bf(v.w * w);
    }
    __syncthreads();

    int wave = tid >> 5;
    int lane = tid & 31;
    constexpr int TN = DOUT / 16;         // col tiles
    int hi = (lane & 16) ? 1 : 0;

    for (int t = wave; t < 4 * TN; t += 8) {
        int tm = t / TN;
        int tn = t % TN;
        v8f acc = {0.f, 0.f, 0.f, 0.f, 0.f, 0.f, 0.f, 0.f};

        #pragma unroll
        for (int kt = 0; kt < 4; ++kt) {
            // A fragment: 16x32 bf16. lanes 0-15: M=lane, K in {0..7,16..23};
            // lanes 16-31: M=lane-16, K in {8..15,24..31}. Two contiguous
            // 16-byte runs per lane -> pairs of ds_load_b128.
            int m  = lane & 15;
            int kb = hi ? 8 : 0;
            v16bf a;
            #pragma unroll
            for (int i = 0; i < 16; ++i) {
                int k = kb + i + ((i & 8) ? 8 : 0);
                a[i] = sA[(tm * 16 + m) * 128 + kt * 32 + k];
            }
            // B fragment: 32x16 bf16. lanes 0-15 hold K in {0..7,16..23}, lanes 16-31 K in {8..15,24..31}.
            int n   = lane & 15;
            int kb2 = kb;
            v16bf b;
            #pragma unroll
            for (int i = 0; i < 16; ++i)
                b[i] = sW[(kt * 32 + kb2 + i + ((i & 8) ? 8 : 0)) * DOUT + tn * 16 + n];

            acc = __builtin_amdgcn_wmma_f32_16x16x32_bf16(
                false, a, false, b, (short)0, acc, false, false);
        }

        // C/D layout: VGPR r -> M = r + 8*hi, N = lane&15
        int col = tn * 16 + (lane & 15);
        float bv = bias[col];
        int rowBase = blockRow + tm * 16 + 8 * hi;
        if (rowBase + 8 <= M) {
            // uniform fast path: 8 unguarded stores (all blocks except the last)
            #pragma unroll
            for (int r = 0; r < 8; ++r) {
                float v = acc[r] + bv;
                out[(size_t)(rowBase + r) * DOUT + col] = v > 0.f ? v : 0.f;
            }
        } else {
            #pragma unroll
            for (int r = 0; r < 8; ++r) {
                int grow = rowBase + r;
                if (grow < M) {
                    float v = acc[r] + bv;
                    out[(size_t)grow * DOUT + col] = v > 0.f ? v : 0.f;
                }
            }
        }
    }
}

// ---------------- per-graph mean pool ----------------
__global__ void k_pool(const float* __restrict__ h, const int* __restrict__ gid,
                       float* gsum, float* gcnt) {
    long idx = (long)blockIdx.x * blockDim.x + threadIdx.x;
    if (idx < (long)NN * DOUTF) {
        int i = (int)(idx >> 6);
        int c = (int)(idx & 63);
        int g = gid[i];
        atomicAdd(&gsum[g * DOUTF + c], h[idx]);
        if (c == 0) atomicAdd(&gcnt[g], 1.0f);
    }
}

__global__ void k_finalize(const float* gsum, const float* gcnt, float* out) {
    int idx = blockIdx.x * blockDim.x + threadIdx.x;
    if (idx < NG * DOUTF) out[idx] = gsum[idx] / fmaxf(gcnt[idx >> 6], 1.0f);
}

// ---------------- launch ----------------
extern "C" void kernel_launch(void* const* d_in, const int* in_sizes, int n_in,
                              void* d_out, int out_size, void* d_ws, size_t ws_size,
                              hipStream_t stream) {
    (void)in_sizes; (void)n_in; (void)out_size; (void)ws_size;
    const float* n_feat = (const float*)d_in[0];
    const float* W0 = (const float*)d_in[1];
    const float* b0 = (const float*)d_in[2];
    const float* W1 = (const float*)d_in[3];
    const float* b1 = (const float*)d_in[4];
    const float* W2 = (const float*)d_in[5];
    const float* b2 = (const float*)d_in[6];
    const int* src  = (const int*)d_in[7];
    const int* dst  = (const int*)d_in[8];
    const int* gid  = (const int*)d_in[9];
    float* out = (float*)d_out;

    char* ws = (char*)d_ws;
    size_t off = 0;
    auto carve = [&](size_t bytes) { char* p = ws + off; off = (off + bytes + 255) & ~(size_t)255; return p; };
    float* normO = (float*)carve((size_t)NN * 4);
    float* normI = (float*)carve((size_t)NN * 4);
    float* agg   = (float*)carve((size_t)NN * 128 * 4);
    float* hA    = (float*)carve((size_t)NN * 128 * 4);
    float* hB    = (float*)carve((size_t)NN * 128 * 4);
    unsigned short* w0b = (unsigned short*)carve((size_t)DI * DH * 2);
    unsigned short* w1b = (unsigned short*)carve((size_t)DH * DH * 2);
    unsigned short* w2b = (unsigned short*)carve((size_t)DH * DOUTF * 2);
    float* gsum = (float*)carve((size_t)NG * DOUTF * 4);
    float* gcnt = (float*)carve((size_t)NG * 4);

    const int T = 256;
    auto blocks = [](long n, int t) { return (int)((n + t - 1) / t); };

    // degrees + norms (normO/normI double as degree accumulators)
    k_zero<<<blocks(NN, T), T, 0, stream>>>(normO, NN);
    k_zero<<<blocks(NN, T), T, 0, stream>>>(normI, NN);
    k_degrees<<<blocks(NE, T), T, 0, stream>>>(src, dst, normO, normI, NE);
    k_norms<<<blocks(NN, T), T, 0, stream>>>(normO, normI, NN);

    // bf16 weights
    k_cvt_bf16<<<blocks(DI * DH, T), T, 0, stream>>>(W0, w0b, DI * DH);
    k_cvt_bf16<<<blocks(DH * DH, T), T, 0, stream>>>(W1, w1b, DH * DH);
    k_cvt_bf16<<<blocks(DH * DOUTF, T), T, 0, stream>>>(W2, w2b, DH * DOUTF);

    const int gemmBlocks = (NN + 63) / 64;

    // layer 0: n_feat -> hA
    k_zero<<<blocks((long)NN * 128, T), T, 0, stream>>>(agg, (long)NN * 128);
    k_scatter<<<(NE + 7) / 8, T, 0, stream>>>(n_feat, normO, src, dst, agg, NE);
    k_gemm_relu<128><<<gemmBlocks, T, 0, stream>>>(agg, normI, w0b, b0, hA, NN);

    // layer 1: hA -> hB
    k_zero<<<blocks((long)NN * 128, T), T, 0, stream>>>(agg, (long)NN * 128);
    k_scatter<<<(NE + 7) / 8, T, 0, stream>>>(hA, normO, src, dst, agg, NE);
    k_gemm_relu<128><<<gemmBlocks, T, 0, stream>>>(agg, normI, w1b, b1, hB, NN);

    // layer 2: hB -> hA (reused, now [NN x 64])
    k_zero<<<blocks((long)NN * 128, T), T, 0, stream>>>(agg, (long)NN * 128);
    k_scatter<<<(NE + 7) / 8, T, 0, stream>>>(hB, normO, src, dst, agg, NE);
    k_gemm_relu<64><<<gemmBlocks, T, 0, stream>>>(agg, normI, w2b, b2, hA, NN);

    // per-graph mean
    k_zero<<<blocks((long)NG * DOUTF, T), T, 0, stream>>>(gsum, (long)NG * DOUTF);
    k_zero<<<blocks(NG, T), T, 0, stream>>>(gcnt, NG);
    k_pool<<<blocks((long)NN * DOUTF, T), T, 0, stream>>>(hA, gid, gsum, gcnt);
    k_finalize<<<blocks((long)NG * DOUTF, T), T, 0, stream>>>(gsum, gcnt, out);
}
